// FMoE_37340445671698
// MI455X (gfx1250) — compile-verified, matrix operand order checked by reading, not run on hardware
//
#include <hip/hip_runtime.h>

#define T_TOK  8192
#define DMODEL 1024
#define HDIM   4096
#define NEXP   16

typedef __bf16 v16bf __attribute__((ext_vector_type(16)));
typedef __bf16 v2bf  __attribute__((ext_vector_type(2)));
typedef float  v8f   __attribute__((ext_vector_type(8)));

union Frag16 { v16bf v; unsigned int u[8]; };

// LDS strides (16-bit units), padded to dodge bank conflicts
#define XS_STR 1032   // 1024 + 8
#define HS_STR 136    // 128 + 8

// fragment K-offset for element-pair p (0..7) and lane-half h (ISA 7.12.2)
#define KOFF(p, h) ((((p) & 3) << 1) + (((p) >> 2) << 4) + ((h) << 3))

__device__ __forceinline__ unsigned short f2bf(float f) {
  union { __bf16 b; unsigned short u; } c;
  c.b = (__bf16)f;   // fptrunc RNE
  return c.u;
}

__device__ __forceinline__ unsigned int pk_bf16(float lo, float hi) {
  union { v2bf v; unsigned int u; } c;
  c.v[0] = (__bf16)lo;
  c.v[1] = (__bf16)hi;
  return c.u;
}

// A-style fragment from LDS (row-major bf16, K pairs contiguous -> dword loads)
__device__ __forceinline__ v16bf load_frag_lds(const unsigned short* base,
                                               int stride_shorts, int row,
                                               int k0, int half) {
  const unsigned int* rp = (const unsigned int*)(base + row * stride_shorts);
  Frag16 f;
#pragma unroll
  for (int p = 0; p < 8; ++p) f.u[p] = rp[(k0 + KOFF(p, half)) >> 1];
  return f.v;
}

// B fragment straight from global fp32: element (K) at p[K*ld], convert+pack.
// Lane's column is baked into p; 16 lanes/half read 16 consecutive floats.
__device__ __forceinline__ v16bf load_b_global(const float* __restrict__ p,
                                               int ld, int half) {
  Frag16 f;
#pragma unroll
  for (int q = 0; q < 8; ++q) {
    int K = KOFF(q, half);
    float lo = p[(size_t)K * ld];
    float hi = p[(size_t)(K + 1) * ld];
    f.u[q] = pk_bf16(lo, hi);
  }
  return f.v;
}

// ---------------- Gating: logits -> top-2 -> softmax -> scatter ----------------
__global__ void __launch_bounds__(256)
moe_gate_kernel(const float* __restrict__ x, const float* __restrict__ Wg,
                const float* __restrict__ bg, int* __restrict__ counts,
                int* __restrict__ tokids, float* __restrict__ gates) {
  const int lane = threadIdx.x & 31;
  const int tok  = blockIdx.x * (blockDim.x >> 5) + (threadIdx.x >> 5);

  float lg[NEXP];
#pragma unroll
  for (int e = 0; e < NEXP; ++e) lg[e] = 0.f;

  const float* xr = x + (size_t)tok * DMODEL;
  for (int d = lane; d < DMODEL; d += 32) {
    float xv = xr[d];
    const float4* wr = (const float4*)(Wg + (size_t)d * NEXP);
#pragma unroll
    for (int q = 0; q < 4; ++q) {
      float4 w4 = wr[q];
      lg[q * 4 + 0] += xv * w4.x;
      lg[q * 4 + 1] += xv * w4.y;
      lg[q * 4 + 2] += xv * w4.z;
      lg[q * 4 + 3] += xv * w4.w;
    }
  }
#pragma unroll
  for (int off = 16; off > 0; off >>= 1)
#pragma unroll
    for (int e = 0; e < NEXP; ++e) lg[e] += __shfl_down(lg[e], off, 32);

  if (lane == 0) {
    int i0 = 0, i1 = 0;
    float v0 = -1e30f, v1 = -1e30f;
#pragma unroll
    for (int e = 0; e < NEXP; ++e) {
      float v = lg[e] + bg[e];
      if (v > v0) { v1 = v0; i1 = i0; v0 = v; i0 = e; }
      else if (v > v1) { v1 = v; i1 = e; }
    }
    float e1  = __expf(v1 - v0);
    float inv = 1.f / (1.f + e1);
    int p0 = atomicAdd(&counts[i0], 1);
    tokids[i0 * T_TOK + p0] = tok;
    gates[i0 * T_TOK + p0] = inv;
    int p1 = atomicAdd(&counts[i1], 1);
    tokids[i1 * T_TOK + p1] = tok;
    gates[i1 * T_TOK + p1] = e1 * inv;
  }
}

// ---------------- Expert FFN: y = relu(X W1 + b1) W2 + b2, gated scatter-add ----------------
__global__ void __launch_bounds__(256)
moe_ffn_kernel(const float* __restrict__ x, const float* __restrict__ W1,
               const float* __restrict__ b1, const float* __restrict__ W2,
               const float* __restrict__ b2, const int* __restrict__ counts,
               const int* __restrict__ tokids, const float* __restrict__ gates,
               float* __restrict__ out) {
  __shared__ unsigned short xs[16 * XS_STR];       // X tile, bf16
  __shared__ unsigned short hs[2 * 16 * HS_STR];   // ping-pong h tile, bf16
  __shared__ int   toks[16];
  __shared__ float gs[16];

  const int e   = blockIdx.y;
  const int m0  = blockIdx.x * 16;
  const int cnt = counts[e];
  if (m0 >= cnt) return;

  const int tid  = threadIdx.x;
  const int lane = tid & 31;
  const int w    = tid >> 5;       // wave 0..7
  const int nidx = lane & 15;
  const int half = lane >> 4;
  const int midx = lane & 15;

  if (tid < 16) {
    bool valid = (m0 + tid) < cnt;
    int slot   = e * T_TOK + (valid ? (m0 + tid) : m0);
    toks[tid]  = tokids[slot];
    gs[tid]    = valid ? gates[slot] : 0.f;
  }
  __syncthreads();

  { // stage X rows fp32 -> bf16 LDS (16 threads/row, float4 coalesced)
    int r  = tid >> 4;
    int c0 = (tid & 15) * 64;
    const float4* xr4 = (const float4*)(x + (size_t)toks[r] * DMODEL);
    unsigned int* dst = (unsigned int*)(xs + r * XS_STR);
#pragma unroll
    for (int j = 0; j < 16; ++j) {
      float4 f = xr4[(c0 >> 2) + j];
      int ci = (c0 + 4 * j) >> 1;
      dst[ci]     = pk_bf16(f.x, f.y);
      dst[ci + 1] = pk_bf16(f.z, f.w);
    }
  }
  __syncthreads();

  const float* W1e = W1 + (size_t)e * DMODEL * HDIM;
  const float* W2e = W2 + (size_t)e * HDIM * DMODEL;

  const v8f vzero = {0.f, 0.f, 0.f, 0.f, 0.f, 0.f, 0.f, 0.f};
  v8f yacc[8];
#pragma unroll
  for (int t = 0; t < 8; ++t) yacc[t] = vzero;

  for (int hc = 0; hc < HDIM; hc += 128) {
    unsigned short* hsb = hs + ((hc >> 7) & 1) * (16 * HS_STR);

    // ---- stage 1: h tile = relu(X @ W1[:, hc:hc+128] + b1)
    // wave w owns h-cols hc+16w .. +15; B fragments direct from global
    v8f hacc = vzero;
    const float* w1col = W1e + hc + w * 16 + nidx;   // + (d0+K)*HDIM
    for (int d0 = 0; d0 < DMODEL; d0 += 32) {
      __builtin_prefetch((const void*)(w1col + (size_t)(d0 + 64) * HDIM), 0, 0);
      v16bf a = load_frag_lds(xs, XS_STR, midx, d0, half);
      v16bf b = load_b_global(w1col + (size_t)d0 * HDIM, HDIM, half);
      hacc = __builtin_amdgcn_wmma_f32_16x16x32_bf16(false, a, false, b,
                                                     (short)0, hacc, false, false);
    }
    { // bias + relu + store h tile (C/D layout: m = v + 8*half, n = lane&15)
      float bv = b1[(size_t)e * HDIM + hc + w * 16 + nidx];
#pragma unroll
      for (int v = 0; v < 8; ++v) {
        float hv = hacc[v] + bv;
        hv = hv > 0.f ? hv : 0.f;
        hsb[(v + 8 * half) * HS_STR + w * 16 + nidx] = f2bf(hv);
      }
    }
    __syncthreads();   // h tile visible to all waves (ping-pong: 1 barrier/chunk)

    // ---- stage 2: y += h @ W2[hc:hc+128, :]; wave w owns out-cols 128w..128w+127
    for (int kk = 0; kk < 128; kk += 32) {
      v16bf a = load_frag_lds(hsb, HS_STR, midx, kk, half);
      const float* w2base = W2e + (size_t)(hc + kk) * DMODEL + w * 128 + nidx;
      __builtin_prefetch((const void*)(w2base + (size_t)32 * DMODEL), 0, 0);
#pragma unroll
      for (int t = 0; t < 8; ++t) {
        v16bf b = load_b_global(w2base + t * 16, DMODEL, half);
        yacc[t] = __builtin_amdgcn_wmma_f32_16x16x32_bf16(false, a, false, b,
                                                          (short)0, yacc[t], false, false);
      }
    }
  }

  // ---- epilogue: out[tok, col] += gate * (y + b2)
  int   tk[8];
  float gg[8];
#pragma unroll
  for (int v = 0; v < 8; ++v) {
    tk[v] = toks[v + 8 * half];
    gg[v] = gs[v + 8 * half];
  }
#pragma unroll
  for (int t = 0; t < 8; ++t) {
    int col = w * 128 + t * 16 + nidx;
    float b2v = b2[(size_t)e * DMODEL + col];
#pragma unroll
    for (int v = 0; v < 8; ++v) {
      float val = (yacc[t][v] + b2v) * gg[v];
      unsafeAtomicAdd(out + (size_t)tk[v] * DMODEL + col, val);
    }
  }
}

extern "C" void kernel_launch(void* const* d_in, const int* in_sizes, int n_in,
                              void* d_out, int out_size, void* d_ws, size_t ws_size,
                              hipStream_t stream) {
  (void)in_sizes; (void)n_in; (void)out_size; (void)ws_size;
  const float* x  = (const float*)d_in[0];
  const float* Wg = (const float*)d_in[1];
  const float* bg = (const float*)d_in[2];
  const float* W1 = (const float*)d_in[3];
  const float* b1 = (const float*)d_in[4];
  const float* W2 = (const float*)d_in[5];
  const float* b2 = (const float*)d_in[6];
  float* out = (float*)d_out;

  // workspace: counts[E] | tokids[E*T] | gates[E*T]  (~1 MB)
  int*   counts = (int*)d_ws;
  int*   tokids = counts + NEXP;
  float* gates  = (float*)(tokids + (size_t)NEXP * T_TOK);

  (void)hipMemsetAsync(counts, 0, NEXP * sizeof(int), stream);
  (void)hipMemsetAsync(out, 0, (size_t)T_TOK * DMODEL * sizeof(float), stream);

  moe_gate_kernel<<<T_TOK / 8, 256, 0, stream>>>(x, Wg, bg, counts, tokids, gates);

  dim3 grid(T_TOK / 16, NEXP);
  moe_ffn_kernel<<<grid, 256, 0, stream>>>(x, W1, b1, W2, b2,
                                           counts, tokids, gates, out);
}
